// QDense_29205777613621
// MI455X (gfx1250) — matrix operand order repaired and checked by reading, not compiled, not run on hardware
//
#include <hip/hip_runtime.h>

// QDense (quaternion dense layer) for MI455X / gfx1250.
//
// y[8192,4096] = x[8192,4096] @ W_block[4096,4096] + b_full
// W_block[k,n] = sgn(kb,nb) * Wsel(kb,nb)[n%1024, k%1024]
//
// Compute-bound split-bf16 (Markidis) GEMM on v_wmma_f32_16x16x32_bf16.
// v3b: hot loop stages LDS tiles with GLOBAL_LOAD_ASYNC_TO_LDS_B128
// (ASYNCcnt-tracked, no staging VGPRs -> no scratch spills) into a
// double-buffered 64KB LDS with XOR-swizzled 16B granules; wmma on buffer A
// overlaps the async fill of buffer B, one barrier per k-step.

#define BM 128
#define BN 128
#define BK 32
#define TILE_U16 (128 * 32)   // one operand tile: 128 rows x 32 bf16
#define LDA 40                // padded stride for the fused fallback kernel
#define LDB 40

typedef unsigned short u16;
typedef __attribute__((ext_vector_type(16))) __bf16 v16bf;
typedef __attribute__((ext_vector_type(8)))  float  v8f;
typedef __attribute__((ext_vector_type(4)))  int    v4i;

#define AS1 __attribute__((address_space(1)))
#define AS3 __attribute__((address_space(3)))

#if __has_builtin(__builtin_amdgcn_global_load_async_to_lds_b128)
#define HAVE_ASYNC_LDS 1
#else
#define HAVE_ASYNC_LDS 0
#endif

// Quaternion block tables: row = kb (input component), col = nb (output component)
__device__ __constant__ int   c_sel[4][4] = {{0,1,2,3},{1,0,3,2},{2,3,0,1},{3,2,1,0}};
__device__ __constant__ float c_sgn[4][4] = {{ 1.f, 1.f, 1.f, 1.f},
                                             {-1.f, 1.f,-1.f, 1.f},
                                             {-1.f, 1.f, 1.f,-1.f},
                                             {-1.f,-1.f, 1.f, 1.f}};

__device__ __forceinline__ unsigned bf_rne(float f) {
  unsigned x = __float_as_uint(f);
  return (x + 0x7FFFu + ((x >> 16) & 1u)) >> 16;
}

__device__ __forceinline__ void split2(float f, unsigned& h, unsigned& l) {
  h = bf_rne(f);
  float fh = __uint_as_float(h << 16);
  l = bf_rne(f - fh);
}

union Frag {
  uint4  q[2];
  v16bf  v;
};

// ---------------------------------------------------------------------------
// Pack kernels (one-time; ~336 MB traffic ~= 15us @ 23.3 TB/s)
// ---------------------------------------------------------------------------

__global__ __launch_bounds__(256) void pack_x_kernel(const float* __restrict__ x,
                                                     u16* __restrict__ Xhi,
                                                     u16* __restrict__ Xlo) {
  const size_t i = ((size_t)blockIdx.x * 256 + threadIdx.x) * 4;
  const float4 f = *(const float4*)(x + i);
  unsigned h[4], l[4];
  const float* fe = (const float*)&f;
#pragma unroll
  for (int e = 0; e < 4; ++e) split2(fe[e], h[e], l[e]);
  *(uint2*)(Xhi + i) = make_uint2(h[0] | (h[1] << 16), h[2] | (h[3] << 16));
  *(uint2*)(Xlo + i) = make_uint2(l[0] | (l[1] << 16), l[2] | (l[3] << 16));
}

// Bt[n][k] = sgn(kb,nb) * Wsel[n%1024][k%1024], n,k in [0,4096)
__global__ __launch_bounds__(256) void pack_w_kernel(
    const float* __restrict__ Wr, const float* __restrict__ Wi,
    const float* __restrict__ Wj, const float* __restrict__ Wk,
    u16* __restrict__ Bhi, u16* __restrict__ Blo) {
  const size_t i = ((size_t)blockIdx.x * 256 + threadIdx.x) * 4;
  const int n = (int)(i >> 12);
  const int k = (int)(i & 4095);
  const int kb = k >> 10, nb = n >> 10;
  const int s = c_sel[kb][nb];
  const float sg = c_sgn[kb][nb];
  const float* __restrict__ W = (s == 0) ? Wr : (s == 1) ? Wi : (s == 2) ? Wj : Wk;
  const float4 f = *(const float4*)(W + (size_t)(n & 1023) * 1024 + (k & 1023));
  unsigned h[4], l[4];
  const float* fe = (const float*)&f;
#pragma unroll
  for (int e = 0; e < 4; ++e) split2(sg * fe[e], h[e], l[e]);
  *(uint2*)(Bhi + i) = make_uint2(h[0] | (h[1] << 16), h[2] | (h[3] << 16));
  *(uint2*)(Blo + i) = make_uint2(l[0] | (l[1] << 16), l[2] | (l[3] << 16));
}

// ---------------------------------------------------------------------------
// Async global->LDS staging (CDNA5 path), 16B-granule XOR swizzle
//   phys offset(row, g) = row*32 + ((g + row/4) & 3) * 8   [u16 units]
// ---------------------------------------------------------------------------

__device__ __forceinline__ void async_cp_b128(u16* lds, const u16* g) {
#if HAVE_ASYNC_LDS
  __builtin_amdgcn_global_load_async_to_lds_b128((AS1 v4i*)g, (AS3 v4i*)lds, 0, 0);
#else
  *(uint4*)lds = *(const uint4*)g;
#endif
}

__device__ __forceinline__ void wait_async_all() {
#if HAVE_ASYNC_LDS
#if __has_builtin(__builtin_amdgcn_s_wait_asynccnt)
  __builtin_amdgcn_s_wait_asynccnt(0);
#else
  asm volatile("s_wait_asynccnt 0x0" ::: "memory");
#endif
#endif
}

// 8 async b128 per thread: 4 operand arrays x 2 row-passes.
__device__ __forceinline__ void issue_tile(
    const u16* __restrict__ Xhi, const u16* __restrict__ Xlo,
    const u16* __restrict__ Bh,  const u16* __restrict__ Bl,
    u16* sbuf, int m0, int n0, int k0, int t) {
  const int row = t >> 2;        // 0..63
  const int g   = t & 3;         // 16B granule within the 32-bf16 row
#pragma unroll
  for (int p = 0; p < 2; ++p) {
    const int rr  = p * 64 + row;
    const int lo  = rr * 32 + (((g + (rr >> 2)) & 3) * 8);
    const size_t ao = (size_t)(m0 + rr) * 4096 + k0 + g * 8;
    const size_t bo = (size_t)(n0 + rr) * 4096 + k0 + g * 8;
    async_cp_b128(sbuf + 0 * TILE_U16 + lo, Xhi + ao);
    async_cp_b128(sbuf + 1 * TILE_U16 + lo, Xlo + ao);
    async_cp_b128(sbuf + 2 * TILE_U16 + lo, Bh + bo);
    async_cp_b128(sbuf + 3 * TILE_U16 + lo, Bl + bo);
  }
}

// ---------------------------------------------------------------------------
// Fragment compute (swizzled layout): 24 wmma per call per wave
// ---------------------------------------------------------------------------

__device__ __forceinline__ void frag_compute_swz(const u16* sbuf,
                                                 int waveM, int waveN, int lm, int lh,
                                                 v8f acc[2][4]) {
  const u16* sAhi = sbuf;
  const u16* sAlo = sbuf + TILE_U16;
  const u16* sBhi = sbuf + 2 * TILE_U16;
  const u16* sBlo = sbuf + 3 * TILE_U16;

  // A (16x32 bf16): lane M=lane%16; K runs = granules lh and lh+2.
  Frag a[2][2];
#pragma unroll
  for (int mt = 0; mt < 2; ++mt) {
    const int r  = waveM * 32 + mt * 16 + lm;
    const int rs = (r >> 2) & 3;
    const int o0 = r * 32 + (((lh + 0 + rs) & 3) * 8);
    const int o1 = r * 32 + (((lh + 2 + rs) & 3) * 8);
    a[mt][0].q[0] = *(const uint4*)&sAhi[o0];
    a[mt][0].q[1] = *(const uint4*)&sAhi[o1];
    a[mt][1].q[0] = *(const uint4*)&sAlo[o0];
    a[mt][1].q[1] = *(const uint4*)&sAlo[o1];
  }
  // B (32x16 bf16): lane N=lane%16; K runs = granules lh and lh+2.
  Frag b[4][2];
#pragma unroll
  for (int nt = 0; nt < 4; ++nt) {
    const int r  = waveN * 64 + nt * 16 + lm;
    const int rs = (r >> 2) & 3;
    const int o0 = r * 32 + (((lh + 0 + rs) & 3) * 8);
    const int o1 = r * 32 + (((lh + 2 + rs) & 3) * 8);
    b[nt][0].q[0] = *(const uint4*)&sBhi[o0];
    b[nt][0].q[1] = *(const uint4*)&sBhi[o1];
    b[nt][1].q[0] = *(const uint4*)&sBlo[o0];
    b[nt][1].q[1] = *(const uint4*)&sBlo[o1];
  }
#pragma unroll
  for (int mt = 0; mt < 2; ++mt)
#pragma unroll
    for (int nt = 0; nt < 4; ++nt) {
      acc[mt][nt] = __builtin_amdgcn_wmma_f32_16x16x32_bf16(
          false, a[mt][0].v, false, b[nt][0].v, (short)0, acc[mt][nt], false, false);
      acc[mt][nt] = __builtin_amdgcn_wmma_f32_16x16x32_bf16(
          false, a[mt][0].v, false, b[nt][1].v, (short)0, acc[mt][nt], false, false);
      acc[mt][nt] = __builtin_amdgcn_wmma_f32_16x16x32_bf16(
          false, a[mt][1].v, false, b[nt][0].v, (short)0, acc[mt][nt], false, false);
    }
}

__global__ __launch_bounds__(256) void qdense_gemm_packed(
    const u16* __restrict__ Xhi, const u16* __restrict__ Xlo,
    const u16* __restrict__ Bh,  const u16* __restrict__ Bl,
    const float* __restrict__ bias, float* __restrict__ out) {
  __shared__ __align__(16) u16 smem[2 * 4 * TILE_U16];  // 64 KB, double-buffered

  const int t  = threadIdx.x;
  const int m0 = blockIdx.y * BM;
  const int n0 = blockIdx.x * BN;

  const int wid   = t >> 5;
  const int lane  = t & 31;
  const int waveM = wid & 3;
  const int waveN = wid >> 2;
  const int lm    = lane & 15;
  const int lh    = lane >> 4;

  v8f acc[2][4];
#pragma unroll
  for (int mt = 0; mt < 2; ++mt)
#pragma unroll
    for (int nt = 0; nt < 4; ++nt) {
      v8f z = {0.f, 0.f, 0.f, 0.f, 0.f, 0.f, 0.f, 0.f};
      acc[mt][nt] = z;
    }

  u16* sb0 = smem;
  u16* sb1 = smem + 4 * TILE_U16;

  issue_tile(Xhi, Xlo, Bh, Bl, sb0, m0, n0, 0, t);

  // 128 k-steps, unrolled by 2 for static buffer selection.
#pragma unroll 1
  for (int kt2 = 0; kt2 < 64; ++kt2) {
    const int k = 2 * kt2 * BK;

    wait_async_all();
    __syncthreads();
    issue_tile(Xhi, Xlo, Bh, Bl, sb1, m0, n0, k + BK, t);
    frag_compute_swz(sb0, waveM, waveN, lm, lh, acc);

    wait_async_all();
    __syncthreads();
    if (kt2 < 63)
      issue_tile(Xhi, Xlo, Bh, Bl, sb0, m0, n0, k + 2 * BK, t);
    frag_compute_swz(sb1, waveM, waveN, lm, lh, acc);
  }

  // Epilogue: bias add + store (C/D layout: N=lane%16, M=v+8*(lane/16)).
#pragma unroll
  for (int nt = 0; nt < 4; ++nt) {
    const int   n  = n0 + waveN * 64 + nt * 16 + lm;
    const float bv = bias[n & 1023];
#pragma unroll
    for (int mt = 0; mt < 2; ++mt) {
      const int mb = m0 + waveM * 32 + mt * 16 + 8 * lh;
#pragma unroll
      for (int v = 0; v < 8; ++v)
        out[(size_t)(mb + v) * 4096 + n] = acc[mt][nt][v] + bv;
    }
  }
}

// ---------------------------------------------------------------------------
// Fallback: fused conversion-in-loop kernel (padded-LDS layout), used only if
// the workspace is too small for the packed operands.
// ---------------------------------------------------------------------------

__device__ __forceinline__ void frag_compute_pad(const u16* sAhi, const u16* sAlo,
                                                 const u16* sBhi, const u16* sBlo,
                                                 int waveM, int waveN, int lm, int lh,
                                                 v8f acc[2][4]) {
  Frag a[2][2];
#pragma unroll
  for (int mt = 0; mt < 2; ++mt) {
    const int off = (waveM * 32 + mt * 16 + lm) * LDA + lh * 8;
    a[mt][0].q[0] = *(const uint4*)&sAhi[off];
    a[mt][0].q[1] = *(const uint4*)&sAhi[off + 16];
    a[mt][1].q[0] = *(const uint4*)&sAlo[off];
    a[mt][1].q[1] = *(const uint4*)&sAlo[off + 16];
  }
  Frag b[4][2];
#pragma unroll
  for (int nt = 0; nt < 4; ++nt) {
    const int off = (waveN * 64 + nt * 16 + lm) * LDB + lh * 8;
    b[nt][0].q[0] = *(const uint4*)&sBhi[off];
    b[nt][0].q[1] = *(const uint4*)&sBhi[off + 16];
    b[nt][1].q[0] = *(const uint4*)&sBlo[off];
    b[nt][1].q[1] = *(const uint4*)&sBlo[off + 16];
  }
#pragma unroll
  for (int mt = 0; mt < 2; ++mt)
#pragma unroll
    for (int nt = 0; nt < 4; ++nt) {
      acc[mt][nt] = __builtin_amdgcn_wmma_f32_16x16x32_bf16(
          false, a[mt][0].v, false, b[nt][0].v, (short)0, acc[mt][nt], false, false);
      acc[mt][nt] = __builtin_amdgcn_wmma_f32_16x16x32_bf16(
          false, a[mt][0].v, false, b[nt][1].v, (short)0, acc[mt][nt], false, false);
      acc[mt][nt] = __builtin_amdgcn_wmma_f32_16x16x32_bf16(
          false, a[mt][1].v, false, b[nt][0].v, (short)0, acc[mt][nt], false, false);
    }
}

__device__ __forceinline__ void gloadA_f(const float* __restrict__ x, int m0, int k0,
                                         int rrow, int kc, float4 r[4]) {
#pragma unroll
  for (int p = 0; p < 4; ++p)
    r[p] = *(const float4*)(x + (size_t)(m0 + p * 32 + rrow) * 4096 + k0 + kc);
}

__device__ __forceinline__ void gloadB_f(const float* __restrict__ Wr, const float* __restrict__ Wi,
                                         const float* __restrict__ Wj, const float* __restrict__ Wk,
                                         int n0, int k0, int rrow, int kc, float4 r[4]) {
  const int kb = k0 >> 10;
  const int s  = c_sel[kb][n0 >> 10];
  const float* __restrict__ W = (s == 0) ? Wr : (s == 1) ? Wi : (s == 2) ? Wj : Wk;
  const int kk = (k0 & 1023) + kc;
#pragma unroll
  for (int p = 0; p < 4; ++p) {
    int nn = (n0 + p * 32 + rrow) & 1023;
    r[p] = *(const float4*)(W + (size_t)nn * 1024 + kk);
  }
}

__global__ __launch_bounds__(256) void qdense_wmma_fused(
    const float* __restrict__ x,  const float* __restrict__ Wr,
    const float* __restrict__ Wi, const float* __restrict__ Wj,
    const float* __restrict__ Wk, const float* __restrict__ bias,
    float* __restrict__ out) {
  __shared__ __align__(16) u16 sAhi[BM * LDA];
  __shared__ __align__(16) u16 sAlo[BM * LDA];
  __shared__ __align__(16) u16 sBhi[BN * LDB];
  __shared__ __align__(16) u16 sBlo[BN * LDB];

  const int t  = threadIdx.x;
  const int m0 = blockIdx.y * BM;
  const int n0 = blockIdx.x * BN;
  const int nbIdx = n0 >> 10;

  const int wid   = t >> 5;
  const int lane  = t & 31;
  const int waveM = wid & 3;
  const int waveN = wid >> 2;
  const int lm    = lane & 15;
  const int lh    = lane >> 4;

  const int rrow = t >> 3;
  const int kc   = (t & 7) << 2;

  v8f acc[2][4];
#pragma unroll
  for (int mt = 0; mt < 2; ++mt)
#pragma unroll
    for (int nt = 0; nt < 4; ++nt) {
      v8f z = {0.f, 0.f, 0.f, 0.f, 0.f, 0.f, 0.f, 0.f};
      acc[mt][nt] = z;
    }

  float4 pa[4], pb[4];
  gloadA_f(x, m0, 0, rrow, kc, pa);
  gloadB_f(Wr, Wi, Wj, Wk, n0, 0, rrow, kc, pb);

#pragma unroll 1
  for (int kt = 0; kt < 4096 / BK; ++kt) {
    const int   k0  = kt * BK;
    const float sgn = c_sgn[k0 >> 10][nbIdx];

#pragma unroll
    for (int p = 0; p < 4; ++p) {
      const int row = p * 32 + rrow;
      unsigned h[4], l[4];
      const float* fa = (const float*)&pa[p];
#pragma unroll
      for (int e = 0; e < 4; ++e) split2(fa[e], h[e], l[e]);
      *(uint2*)&sAhi[row * LDA + kc] = make_uint2(h[0] | (h[1] << 16), h[2] | (h[3] << 16));
      *(uint2*)&sAlo[row * LDA + kc] = make_uint2(l[0] | (l[1] << 16), l[2] | (l[3] << 16));
      const float* fb = (const float*)&pb[p];
#pragma unroll
      for (int e = 0; e < 4; ++e) split2(sgn * fb[e], h[e], l[e]);
      *(uint2*)&sBhi[row * LDB + kc] = make_uint2(h[0] | (h[1] << 16), h[2] | (h[3] << 16));
      *(uint2*)&sBlo[row * LDB + kc] = make_uint2(l[0] | (l[1] << 16), l[2] | (l[3] << 16));
    }
    __syncthreads();

    float4 na[4], nb[4];
    const bool more = (kt + 1) < (4096 / BK);
    if (more) {
      gloadA_f(x, m0, k0 + BK, rrow, kc, na);
      gloadB_f(Wr, Wi, Wj, Wk, n0, k0 + BK, rrow, kc, nb);
    }

    frag_compute_pad(sAhi, sAlo, sBhi, sBlo, waveM, waveN, lm, lh, acc);

    __syncthreads();
    if (more) {
#pragma unroll
      for (int p = 0; p < 4; ++p) { pa[p] = na[p]; pb[p] = nb[p]; }
    }
  }

#pragma unroll
  for (int nt = 0; nt < 4; ++nt) {
    const int   n  = n0 + waveN * 64 + nt * 16 + lm;
    const float bv = bias[n & 1023];
#pragma unroll
    for (int mt = 0; mt < 2; ++mt) {
      const int mb = m0 + waveM * 32 + mt * 16 + 8 * lh;
#pragma unroll
      for (int v = 0; v < 8; ++v)
        out[(size_t)(mb + v) * 4096 + n] = acc[mt][nt][v] + bv;
    }
  }
}

// ---------------------------------------------------------------------------

extern "C" void kernel_launch(void* const* d_in, const int* in_sizes, int n_in,
                              void* d_out, int out_size, void* d_ws, size_t ws_size,
                              hipStream_t stream) {
  (void)in_sizes; (void)n_in; (void)out_size;
  const float* x  = (const float*)d_in[0];
  const float* Wr = (const float*)d_in[1];
  const float* Wi = (const float*)d_in[2];
  const float* Wj = (const float*)d_in[3];
  const float* Wk = (const float*)d_in[4];
  const float* b  = (const float*)d_in[5];
  float* out = (float*)d_out;

  const size_t xElems = (size_t)8192 * 4096;
  const size_t wElems = (size_t)4096 * 4096;
  const size_t need   = (2 * xElems + 2 * wElems) * sizeof(u16);  // 192 MiB

  dim3 grid(4096 / BN, 8192 / BM);  // 32 x 64 = 2048 workgroups
  dim3 block(256);                  // 8 wave32s

  if (ws_size >= need && d_ws != nullptr) {
    u16* Xhi = (u16*)d_ws;
    u16* Xlo = Xhi + xElems;
    u16* Bh  = Xlo + xElems;
    u16* Bl  = Bh + wElems;
    pack_x_kernel<<<(int)(xElems / (256 * 4)), 256, 0, stream>>>(x, Xhi, Xlo);
    pack_w_kernel<<<(int)(wElems / (256 * 4)), 256, 0, stream>>>(Wr, Wi, Wj, Wk, Bh, Bl);
    qdense_gemm_packed<<<grid, block, 0, stream>>>(Xhi, Xlo, Bh, Bl, b, out);
  } else {
    qdense_wmma_fused<<<grid, block, 0, stream>>>(x, Wr, Wi, Wj, Wk, b, out);
  }
}
